// WorldModel_89352499626372
// MI455X (gfx1250) — matrix operand order repaired and checked by reading, not compiled, not run on hardware
//
#include <hip/hip_runtime.h>
#include <hip/hip_bf16.h>
#include <math.h>

// ---------------------------------------------------------------------------
// Types for CDNA5 WMMA (wave32): bf16 16x16x32 -> f32 accumulator
// ---------------------------------------------------------------------------
typedef __bf16 bf16_t;
typedef __attribute__((ext_vector_type(16))) bf16_t v16bf;
typedef __attribute__((ext_vector_type(8)))  bf16_t v8bf;
typedef __attribute__((ext_vector_type(8)))  float  v8f;

__device__ __forceinline__ v16bf cat8(v8bf lo, v8bf hi) {
  return __builtin_shufflevector(lo, hi, 0,1,2,3,4,5,6,7,8,9,10,11,12,13,14,15);
}

__device__ __forceinline__ v8f wmma_bf16(v16bf a, v16bf b, v8f c) {
  // (neg_a, A, neg_b, B, c_mod, C, reuse_a, reuse_b)
  return __builtin_amdgcn_wmma_f32_16x16x32_bf16(false, a, false, b, (short)0, c,
                                                 false, false);
}

__device__ __forceinline__ float elu_f(float x) {
  return x > 0.0f ? x : (expf(x) - 1.0f);
}
__device__ __forceinline__ float softplus_f(float x) {
  return x > 20.0f ? x : logf(1.0f + expf(x));
}
__device__ __forceinline__ float sigmoid_f(float x) {
  return 1.0f / (1.0f + expf(-x));
}

// ---------------------------------------------------------------------------
// fp32 -> bf16 conversion
// ---------------------------------------------------------------------------
__global__ void cvt_f32_bf16(const float* __restrict__ src, bf16_t* __restrict__ dst,
                             size_t n) {
  size_t i = (size_t)blockIdx.x * blockDim.x + threadIdx.x;
  size_t stride = (size_t)gridDim.x * blockDim.x;
  for (; i < n; i += stride) dst[i] = (bf16_t)src[i];
}

// ---------------------------------------------------------------------------
// Generic WMMA GEMM:  C[M,N] = epi( A[M,K](bf16,row-major,lda) * B )
// where B[k][n] = W[n0+n][colofs+k], W row-major bf16 with row stride ldw.
// One 16x16 C tile per wave. mode: 0 = raw f32 out (ld=N)
//                                  1 = +bias, ELU, bf16 out (ld=N)
//                                  2 = +bias, f32 out with col<Nstore guard (ld=Nstore)
// ---------------------------------------------------------------------------
__global__ __launch_bounds__(128)
void gemm_bt_wmma(const bf16_t* __restrict__ A, int lda,
                  const bf16_t* __restrict__ W, int ldw, int colofs,
                  const float* __restrict__ bias,
                  float* __restrict__ outF, bf16_t* __restrict__ outB,
                  int ntiles, int ntilesN, int N, int Nstore, int K, int mode) {
  int wid = blockIdx.x * (blockDim.x >> 5) + (threadIdx.x >> 5);
  if (wid >= ntiles) return;                 // wave-uniform
  int lane = threadIdx.x & 31;
  int hi   = (lane >= 16);
  int r    = lane & 15;
  int m0 = (wid / ntilesN) * 16;
  int n0 = (wid % ntilesN) * 16;

  const bf16_t* ap = A + (size_t)(m0 + r) * lda + (hi ? 8 : 0);
  const bf16_t* bp = W + (size_t)(n0 + r) * ldw + colofs + (hi ? 8 : 0);

  v8f acc = {};
  for (int k = 0; k < K; k += 32) {
    v8bf a0 = *(const v8bf*)(ap + k);
    v8bf a1 = *(const v8bf*)(ap + k + 16);
    v8bf b0 = *(const v8bf*)(bp + k);
    v8bf b1 = *(const v8bf*)(bp + k + 16);
    acc = wmma_bf16(cat8(a0, a1), cat8(b0, b1), acc);
  }

  int col   = n0 + r;
  int rbase = m0 + (hi ? 8 : 0);
  if (mode == 0) {
#pragma unroll
    for (int i = 0; i < 8; ++i)
      outF[(size_t)(rbase + i) * N + col] = acc[i];
  } else if (mode == 1) {
    float bb = bias[col];
#pragma unroll
    for (int i = 0; i < 8; ++i) {
      float v = elu_f(acc[i] + bb);
      outB[(size_t)(rbase + i) * N + col] = (bf16_t)v;
    }
  } else {
    if (col < Nstore) {
      float bb = bias[col];
#pragma unroll
      for (int i = 0; i < 8; ++i)
        outF[(size_t)(rbase + i) * Nstore + col] = acc[i] + bb;
    }
  }
}

// ---------------------------------------------------------------------------
// WMMA mat-vec: 16 output rows per wave. A = 16x32 weight tile, B = x vector
// broadcast into all 16 columns -> every lane in each half holds the result.
// ---------------------------------------------------------------------------
__device__ __forceinline__ v8f matvec16(const bf16_t* __restrict__ W, int ld, int row0,
                                        const bf16_t* __restrict__ x, int K, int lane) {
  int hi = (lane >= 16);
  const bf16_t* wp = W + (size_t)(row0 + (lane & 15)) * ld + (hi ? 8 : 0);
  const bf16_t* xp = x + (hi ? 8 : 0);
  v8f acc = {};
  for (int k = 0; k < K; k += 32) {
    v8bf a0 = *(const v8bf*)(wp + k);
    v8bf a1 = *(const v8bf*)(wp + k + 16);
    v8bf b0 = *(const v8bf*)(xp + k);
    v8bf b1 = *(const v8bf*)(xp + k + 16);
    acc = wmma_bf16(cat8(a0, a1), cat8(b0, b1), acc);
  }
  return acc;
}

// ---------------------------------------------------------------------------
// Persistent recurrent kernel over T steps with device-wide barriers
// ---------------------------------------------------------------------------
struct StepParams {
  const bf16_t *p_enc_w, *gru_wih, *gru_whh, *p_dec1_w, *p_dec2_w, *q_enc_w, *q_dec_w;
  const float  *p_enc_b, *gru_b, *gru_bn, *p_dec1_b, *p_dec2_b, *q_enc_b, *q_dec_b;
  const float  *pa, *qe, *noise;
  bf16_t *x_bf, *det_bf, *stoch_bf, *y1_bf, *q_bf, *states_bf;
  float  *ig, *hg, *det_f32, *y2, *qv;
  float  *o_stoch, *o_det, *o_qmu, *o_qstd, *o_pmu, *o_pstd;
  unsigned *bar_cnt;
  unsigned *bar_gen;
  unsigned nwg;
  int T;
};

__device__ __forceinline__ void grid_barrier(unsigned* cnt, volatile unsigned* gen,
                                             unsigned nwg) {
  __threadfence();
  __syncthreads();
  if (threadIdx.x == 0) {
    unsigned g = *gen;
    if (atomicAdd(cnt, 1u) == nwg - 1u) {
      atomicExch(cnt, 0u);
      __threadfence();
      atomicExch((unsigned*)gen, g + 1u);
    } else {
      while (*gen == g) { __builtin_amdgcn_s_sleep(1); }
    }
    __threadfence();
  }
  __syncthreads();
}

#define DD 2048
#define SS 1024

__global__ __launch_bounds__(256)
void rssm_scan_kernel(StepParams P) {
  const int lane = threadIdx.x & 31;
  const int wid = (blockIdx.x * blockDim.x + threadIdx.x) >> 5;
  const int nwaves = (gridDim.x * blockDim.x) >> 5;
  const int tid = blockIdx.x * blockDim.x + threadIdx.x;
  const int nthreads = gridDim.x * blockDim.x;
  const bool storer = (lane == 0) || (lane == 16);
  const int rofs = (lane == 16) ? 8 : 0;

  // init carry = 0
  for (int i = tid; i < DD; i += nthreads) {
    P.det_bf[i] = (bf16_t)0.0f;
    P.det_f32[i] = 0.0f;
    if (i < SS) P.stoch_bf[i] = (bf16_t)0.0f;
  }
  grid_barrier(P.bar_cnt, P.bar_gen, P.nwg);

  for (int t = 0; t < P.T; ++t) {
    const float* pa_t = P.pa + (size_t)t * DD;
    const float* qe_t = P.qe + (size_t)t * DD;

    // ---- Stage A: x = elu(p_enc_w[:, :S] @ stoch_prev + pa[t] + p_enc_b)
    //               hg = gru_whh @ det_prev                (independent)
    for (int tile = wid; tile < 128 + 384; tile += nwaves) {
      if (tile < 128) {
        int row0 = tile * 16;
        v8f acc = matvec16(P.p_enc_w, 1088, row0, P.stoch_bf, SS, lane);
        if (storer) {
          int rb = row0 + rofs;
#pragma unroll
          for (int i = 0; i < 8; ++i) {
            float v = elu_f(acc[i] + pa_t[rb + i] + P.p_enc_b[rb + i]);
            P.x_bf[rb + i] = (bf16_t)v;
          }
        }
      } else {
        int row0 = (tile - 128) * 16;
        v8f acc = matvec16(P.gru_whh, DD, row0, P.det_bf, DD, lane);
        if (storer) {
          int rb = row0 + rofs;
#pragma unroll
          for (int i = 0; i < 8; ++i) P.hg[rb + i] = acc[i];
        }
      }
    }
    grid_barrier(P.bar_cnt, P.bar_gen, P.nwg);

    // ---- Stage B: ig = gru_wih @ x + gru_b
    for (int tile = wid; tile < 384; tile += nwaves) {
      int row0 = tile * 16;
      v8f acc = matvec16(P.gru_wih, DD, row0, P.x_bf, DD, lane);
      if (storer) {
        int rb = row0 + rofs;
#pragma unroll
        for (int i = 0; i < 8; ++i) P.ig[rb + i] = acc[i] + P.gru_b[rb + i];
      }
    }
    grid_barrier(P.bar_cnt, P.bar_gen, P.nwg);

    // ---- Stage C: GRU combine (elementwise, in-place carry update)
    if (tid < DD) {
      int i = tid;
      float r = sigmoid_f(P.ig[i] + P.hg[i]);
      float z = sigmoid_f(P.ig[DD + i] + P.hg[DD + i]);
      float n = tanhf(P.ig[2 * DD + i] + r * (P.hg[2 * DD + i] + P.gru_bn[i]));
      float det = n + z * (P.det_f32[i] - n);
      P.det_f32[i] = det;
      P.det_bf[i] = (bf16_t)det;
      P.o_det[(size_t)t * DD + i] = det;
      P.states_bf[(size_t)t * 3072 + 1024 + i] = (bf16_t)det;
    }
    grid_barrier(P.bar_cnt, P.bar_gen, P.nwg);

    // ---- Stage D: y1 = elu(p_dec1_w @ det + b) ; q = elu(q_enc_w[:, :D] @ det + qe[t] + b)
    for (int tile = wid; tile < 256; tile += nwaves) {
      if (tile < 128) {
        int row0 = tile * 16;
        v8f acc = matvec16(P.p_dec1_w, DD, row0, P.det_bf, DD, lane);
        if (storer) {
          int rb = row0 + rofs;
#pragma unroll
          for (int i = 0; i < 8; ++i)
            P.y1_bf[rb + i] = (bf16_t)elu_f(acc[i] + P.p_dec1_b[rb + i]);
        }
      } else {
        int row0 = (tile - 128) * 16;
        v8f acc = matvec16(P.q_enc_w, 4096, row0, P.det_bf, DD, lane);
        if (storer) {
          int rb = row0 + rofs;
#pragma unroll
          for (int i = 0; i < 8; ++i)
            P.q_bf[rb + i] = (bf16_t)elu_f(acc[i] + qe_t[rb + i] + P.q_enc_b[rb + i]);
        }
      }
    }
    grid_barrier(P.bar_cnt, P.bar_gen, P.nwg);

    // ---- Stage E: y2 = p_dec2_w @ y1 + b ; qv = q_dec_w @ q + b
    for (int tile = wid; tile < 256; tile += nwaves) {
      if (tile < 128) {
        int row0 = tile * 16;
        v8f acc = matvec16(P.p_dec2_w, DD, row0, P.y1_bf, DD, lane);
        if (storer) {
          int rb = row0 + rofs;
#pragma unroll
          for (int i = 0; i < 8; ++i) P.y2[rb + i] = acc[i] + P.p_dec2_b[rb + i];
        }
      } else {
        int row0 = (tile - 128) * 16;
        v8f acc = matvec16(P.q_dec_w, DD, row0, P.q_bf, DD, lane);
        if (storer) {
          int rb = row0 + rofs;
#pragma unroll
          for (int i = 0; i < 8; ++i) P.qv[rb + i] = acc[i] + P.q_dec_b[rb + i];
        }
      }
    }
    grid_barrier(P.bar_cnt, P.bar_gen, P.nwg);

    // ---- Stage F: split mu/std, softplus, reparameterized sample
    if (tid < DD) {
      int i = tid;
      if (i < SS) {
        P.o_pmu[(size_t)t * SS + i] = P.y2[i];
        P.o_qmu[(size_t)t * SS + i] = P.qv[i];
      } else {
        int j = i - SS;
        float ps = softplus_f(P.y2[i]) + 0.1f;
        float qs = softplus_f(P.qv[i]) + 0.1f;
        P.o_pstd[(size_t)t * SS + j] = ps;
        P.o_qstd[(size_t)t * SS + j] = qs;
        float st = P.qv[j] + qs * P.noise[(size_t)t * SS + j];
        P.o_stoch[(size_t)t * SS + j] = st;
        P.stoch_bf[j] = (bf16_t)st;
        P.states_bf[(size_t)t * 3072 + j] = (bf16_t)st;
      }
    }
    grid_barrier(P.bar_cnt, P.bar_gen, P.nwg);
  }
}

// ---------------------------------------------------------------------------
// Host launch
// ---------------------------------------------------------------------------
extern "C" void kernel_launch(void* const* d_in, const int* in_sizes, int n_in,
                              void* d_out, int out_size, void* d_ws, size_t ws_size,
                              hipStream_t stream) {
  (void)in_sizes; (void)n_in; (void)out_size; (void)ws_size;
  const int T = 1024;

  const float* observation = (const float*)d_in[0];
  const float* actions     = (const float*)d_in[1];
  const float* noise       = (const float*)d_in[2];
  const float* enc_w   = (const float*)d_in[3];
  const float* enc_b   = (const float*)d_in[4];
  const float* dec_w   = (const float*)d_in[5];
  const float* dec_b   = (const float*)d_in[6];
  const float* p_enc_w = (const float*)d_in[7];
  const float* p_enc_b = (const float*)d_in[8];
  const float* gru_wih = (const float*)d_in[9];
  const float* gru_whh = (const float*)d_in[10];
  const float* gru_b   = (const float*)d_in[11];
  const float* gru_bn  = (const float*)d_in[12];
  const float* p_dec1_w = (const float*)d_in[13];
  const float* p_dec1_b = (const float*)d_in[14];
  const float* p_dec2_w = (const float*)d_in[15];
  const float* p_dec2_b = (const float*)d_in[16];
  const float* q_enc_w = (const float*)d_in[17];
  const float* q_enc_b = (const float*)d_in[18];
  const float* q_dec_w = (const float*)d_in[19];
  const float* q_dec_b = (const float*)d_in[20];

  float* out = (float*)d_out;
  float* o_stoch = out;                       // T*1024
  float* o_det   = out + 1048576;             // T*2048
  float* o_outs  = out + 3145728;             // T*1026
  float* o_qmu   = out + 4196352;
  float* o_qstd  = out + 5244928;
  float* o_pmu   = out + 6293504;
  float* o_pstd  = out + 7342080;

  // ---- workspace layout
  char* ws = (char*)d_ws;
  size_t off = 0;
  auto alloc = [&](size_t bytes) -> char* {
    char* p = ws + off;
    off = (off + bytes + 255) & ~(size_t)255;
    return p;
  };
  bf16_t* enc_w_bf    = (bf16_t*)alloc((size_t)2048 * 1024 * 2);
  bf16_t* p_enc_w_bf  = (bf16_t*)alloc((size_t)2048 * 1088 * 2);
  bf16_t* gru_wih_bf  = (bf16_t*)alloc((size_t)6144 * 2048 * 2);
  bf16_t* gru_whh_bf  = (bf16_t*)alloc((size_t)6144 * 2048 * 2);
  bf16_t* p_dec1_w_bf = (bf16_t*)alloc((size_t)2048 * 2048 * 2);
  bf16_t* p_dec2_w_bf = (bf16_t*)alloc((size_t)2048 * 2048 * 2);
  bf16_t* q_enc_w_bf  = (bf16_t*)alloc((size_t)2048 * 4096 * 2);
  bf16_t* q_dec_w_bf  = (bf16_t*)alloc((size_t)2048 * 2048 * 2);
  bf16_t* dec_w_bf    = (bf16_t*)alloc((size_t)1040 * 3072 * 2);  // padded rows
  bf16_t* obs_bf      = (bf16_t*)alloc((size_t)1024 * 1024 * 2);
  bf16_t* act_bf      = (bf16_t*)alloc((size_t)1024 * 64 * 2);
  bf16_t* emb_bf      = (bf16_t*)alloc((size_t)1024 * 2048 * 2);
  bf16_t* states_bf   = (bf16_t*)alloc((size_t)1024 * 3072 * 2);
  bf16_t* x_bf     = (bf16_t*)alloc(2048 * 2);
  bf16_t* det_bf   = (bf16_t*)alloc(2048 * 2);
  bf16_t* stoch_bf = (bf16_t*)alloc(1024 * 2);
  bf16_t* y1_bf    = (bf16_t*)alloc(2048 * 2);
  bf16_t* q_bf     = (bf16_t*)alloc(2048 * 2);
  float* pa      = (float*)alloc((size_t)1024 * 2048 * 4);
  float* qe      = (float*)alloc((size_t)1024 * 2048 * 4);
  float* ig      = (float*)alloc(6144 * 4);
  float* hg      = (float*)alloc(6144 * 4);
  float* det_f32 = (float*)alloc(2048 * 4);
  float* y2      = (float*)alloc(2048 * 4);
  float* qv      = (float*)alloc(2048 * 4);
  unsigned* bars = (unsigned*)alloc(2 * 4);

  // zero the barrier + dec_w padding (capture-legal async memset)
  hipMemsetAsync(bars, 0, 2 * sizeof(unsigned), stream);
  hipMemsetAsync(dec_w_bf, 0, (size_t)1040 * 3072 * 2, stream);

  // ---- fp32 -> bf16 conversions
  auto cvt = [&](const float* s, bf16_t* d, size_t n) {
    int blocks = (int)((n + 255) / 256);
    if (blocks > 2048) blocks = 2048;
    cvt_f32_bf16<<<blocks, 256, 0, stream>>>(s, d, n);
  };
  cvt(enc_w,    enc_w_bf,    (size_t)2048 * 1024);
  cvt(p_enc_w,  p_enc_w_bf,  (size_t)2048 * 1088);
  cvt(gru_wih,  gru_wih_bf,  (size_t)6144 * 2048);
  cvt(gru_whh,  gru_whh_bf,  (size_t)6144 * 2048);
  cvt(p_dec1_w, p_dec1_w_bf, (size_t)2048 * 2048);
  cvt(p_dec2_w, p_dec2_w_bf, (size_t)2048 * 2048);
  cvt(q_enc_w,  q_enc_w_bf,  (size_t)2048 * 4096);
  cvt(q_dec_w,  q_dec_w_bf,  (size_t)2048 * 2048);
  cvt(dec_w,    dec_w_bf,    (size_t)1026 * 3072);
  cvt(observation, obs_bf,   (size_t)1024 * 1024);
  cvt(actions,  act_bf,      (size_t)1024 * 64);

  // ---- batch GEMMs (WMMA) -----------------------------------------------
  // emb = elu(obs @ enc_w^T + enc_b) -> bf16 [T,2048]
  {
    int ntN = 2048 / 16, nt = (1024 / 16) * ntN;
    gemm_bt_wmma<<<(nt + 3) / 4, 128, 0, stream>>>(
        obs_bf, 1024, enc_w_bf, 1024, 0, enc_b, nullptr, emb_bf,
        nt, ntN, 2048, 2048, 1024, 1);
  }
  // pa[t] = p_enc_w[:, S:] @ act_t -> f32 [T,2048]
  {
    int ntN = 2048 / 16, nt = (1024 / 16) * ntN;
    gemm_bt_wmma<<<(nt + 3) / 4, 128, 0, stream>>>(
        act_bf, 64, p_enc_w_bf, 1088, 1024, nullptr, pa, nullptr,
        nt, ntN, 2048, 2048, 64, 0);
  }
  // qe[t] = q_enc_w[:, D:] @ emb_t -> f32 [T,2048]
  {
    int ntN = 2048 / 16, nt = (1024 / 16) * ntN;
    gemm_bt_wmma<<<(nt + 3) / 4, 128, 0, stream>>>(
        emb_bf, 2048, q_enc_w_bf, 4096, 2048, nullptr, qe, nullptr,
        nt, ntN, 2048, 2048, 2048, 0);
  }

  // ---- persistent recurrent scan ----------------------------------------
  StepParams P;
  P.p_enc_w = p_enc_w_bf; P.gru_wih = gru_wih_bf; P.gru_whh = gru_whh_bf;
  P.p_dec1_w = p_dec1_w_bf; P.p_dec2_w = p_dec2_w_bf;
  P.q_enc_w = q_enc_w_bf; P.q_dec_w = q_dec_w_bf;
  P.p_enc_b = p_enc_b; P.gru_b = gru_b; P.gru_bn = gru_bn;
  P.p_dec1_b = p_dec1_b; P.p_dec2_b = p_dec2_b;
  P.q_enc_b = q_enc_b; P.q_dec_b = q_dec_b;
  P.pa = pa; P.qe = qe; P.noise = noise;
  P.x_bf = x_bf; P.det_bf = det_bf; P.stoch_bf = stoch_bf;
  P.y1_bf = y1_bf; P.q_bf = q_bf; P.states_bf = states_bf;
  P.ig = ig; P.hg = hg; P.det_f32 = det_f32; P.y2 = y2; P.qv = qv;
  P.o_stoch = o_stoch; P.o_det = o_det; P.o_qmu = o_qmu; P.o_qstd = o_qstd;
  P.o_pmu = o_pmu; P.o_pstd = o_pstd;
  P.bar_cnt = bars; P.bar_gen = bars + 1; P.nwg = 64; P.T = T;
  rssm_scan_kernel<<<64, 256, 0, stream>>>(P);

  // ---- final GEMM: outs = states @ dec_w^T + dec_b (N=1026, pad to 1040)
  {
    int ntN = 1040 / 16, nt = (1024 / 16) * ntN;
    gemm_bt_wmma<<<(nt + 3) / 4, 128, 0, stream>>>(
        states_bf, 3072, dec_w_bf, 3072, 0, dec_b, o_outs, nullptr,
        nt, ntN, 1040, 1026, 3072, 2);
  }
}
